// AttentionClassifier_575525618182
// MI455X (gfx1250) — compile-verified, matrix operand order checked by reading, not run on hardware
//
#include <hip/hip_runtime.h>
#include <hip/hip_bf16.h>

// Dims (fixed by reference)
#define Ldim 128
#define Bdim 32
#define Edim 128
#define Hdim 256
#define NHdim 8
#define ETAdim 100

typedef __attribute__((ext_vector_type(8)))  __bf16 v8bf;
typedef __attribute__((ext_vector_type(16))) __bf16 v16bf;
typedef __attribute__((ext_vector_type(8)))  float  v8f;

struct BfPair { v8bf lo, hi; };
union FragU { v16bf v; BfPair p; };

static __device__ __forceinline__ unsigned short f2bfu(float f) {
  union { float f; unsigned u; } a; a.f = f;
  unsigned r = a.u + 0x7fffu + ((a.u >> 16) & 1u);
  return (unsigned short)(r >> 16);
}
static __device__ __forceinline__ __bf16 f2bf(float f) {
  union { unsigned short s; __bf16 b; } o; o.s = f2bfu(f);
  return o.b;
}
static __device__ __forceinline__ float sigmf(float x) {
  return 1.f / (1.f + expf(-x));
}

// A-fragment (16x32 bf16) from row-major [rows][stride] memory (LDS or global).
// ISA 16-bit A layout: lanes 0-15 rows 0-15 hold K {k0..k0+7, k0+16..k0+23};
// lanes 16-31 same rows hold K {k0+8..k0+15, k0+24..k0+31}.
static __device__ __forceinline__ v16bf load_a_frag(const __bf16* base, int stride,
                                                    int row_base, int k0, int lane) {
  const __bf16* p = base + (size_t)(row_base + (lane & 15)) * stride
                         + (k0 + ((lane >> 4) << 3));
  FragU u;
  u.p.lo = *(const v8bf*)p;
  u.p.hi = *(const v8bf*)(p + 16);
  return u.v;
}

// B-fragment from pre-packed fragment-major buffer: each (kt, nt) tile stores
// 32 lanes x 16 bf16 contiguously, already in WMMA B layout.
static __device__ __forceinline__ v16bf load_b_frag(const __bf16* packed, int ntiles,
                                                    int kt, int nt, int lane) {
  return *(const v16bf*)(packed + ((((size_t)kt * ntiles + nt) * 32 + lane) << 4));
}

static __device__ __forceinline__ v8f wmma_bf(v16bf a, v16bf b, v8f c) {
  // D = A(16x32 bf16) x B(32x16 bf16) + C(16x16 f32)
  return __builtin_amdgcn_wmma_f32_16x16x32_bf16(false, a, false, b, (short)0, c,
                                                 false, false);
}

// CDNA5 async global->LDS copy, 16B per lane, tracked by ASYNCcnt.
static __device__ __forceinline__ void async_g2l_b128(unsigned lds_byte_addr,
                                                      unsigned long long gaddr) {
  asm volatile("global_load_async_to_lds_b128 %0, %1, off"
               :: "v"(lds_byte_addr), "v"(gaddr) : "memory");
}
static __device__ __forceinline__ void wait_asynccnt0() {
  asm volatile("s_wait_asynccnt 0x0" ::: "memory");
}

// ---------------------------------------------------------------------------
// Prep: pack a row-major f32 weight W[n][k] into fragment-major bf16 B layout
// (B[k][n] = W[n][k]), zero-padding out-of-range k/n.
// ---------------------------------------------------------------------------
__global__ void pack_b(const float* __restrict__ W, __bf16* __restrict__ dst,
                       int rowlen, int k_off, int Kvalid, int Nvalid,
                       int ktiles, int ntiles_loc, int nt_off, int ntiles_tot) {
  int idx = blockIdx.x * blockDim.x + threadIdx.x;
  int total = ktiles * ntiles_loc * 512;
  if (idx >= total) return;
  int e    = idx & 15;
  int lane = (idx >> 4) & 31;
  int tile = idx >> 9;
  int nt = tile % ntiles_loc, kt = tile / ntiles_loc;
  int n = nt * 16 + (lane & 15);
  int k = kt * 32 + ((lane >> 4) << 3) + ((e >> 3) << 4) + (e & 7);
  float v = (k < Kvalid && n < Nvalid) ? W[(size_t)n * rowlen + k_off + k] : 0.f;
  size_t di = (((((size_t)kt * ntiles_tot) + nt_off + nt) * 32 + lane) << 4) + e;
  dst[di] = f2bf(v);
}

// ---------------------------------------------------------------------------
// Embedding gather -> bf16 e[L*B][E]
// ---------------------------------------------------------------------------
__global__ void embed_gather(const int* __restrict__ node,
                             const float* __restrict__ table,
                             __bf16* __restrict__ eb) {
  int idx = blockIdx.x * blockDim.x + threadIdx.x; // L*B*E = 524288 exact
  int k  = idx & (Edim - 1);
  int lb = idx >> 7;
  int id = node[lb];
  eb[idx] = f2bf(table[(size_t)id * Edim + k]);
}

// ---------------------------------------------------------------------------
// PQ GEMM: PQ[m][0:128]=P=e@Wa0L.T, PQ[m][128:256]=Q=e@Wa0R.T ; m = l*B+b
// One wave per 16x16 tile; K=128 -> 4 WMMA per tile.
// ---------------------------------------------------------------------------
__global__ __launch_bounds__(256) void gemm_pq(const __bf16* __restrict__ eb,
                                               const __bf16* __restrict__ wa0p,
                                               float* __restrict__ PQ) {
  int lane = threadIdx.x & 31;
  int wid  = (blockIdx.x * blockDim.x + threadIdx.x) >> 5; // 0..4095
  int mt = wid >> 4, nt = wid & 15;
  v8f acc = {};
#pragma unroll
  for (int kt = 0; kt < 4; ++kt) {
    v16bf a = load_a_frag(eb, Edim, mt * 16, kt * 32, lane);
    v16bf b = load_b_frag(wa0p, 16, kt, nt, lane);
    acc = wmma_bf(a, b, acc);
  }
  int n = nt * 16 + (lane & 15);
  int rbase = mt * 16 + ((lane >> 4) << 3);
#pragma unroll
  for (int v = 0; v < 8; ++v)
    PQ[(size_t)(rbase + v) * 256 + n] = acc[v];
}

// ---------------------------------------------------------------------------
// Attention weights: w[i,b,nh] = sum_j relu(P[j]+Q[i]+ba0) . Wa1[nh] + 128*ba1
// One 32-thread block per (b, i-tile of 16). K=(j,k) streamed in 32-chunks.
// ---------------------------------------------------------------------------
__global__ __launch_bounds__(32) void attn_w(const float* __restrict__ PQ,
                                             const float* __restrict__ ba0,
                                             const float* __restrict__ ba1,
                                             const __bf16* __restrict__ wa1p,
                                             float* __restrict__ wout) {
  __shared__ __align__(16) float  Qt[16][128];
  __shared__ __align__(16) __bf16 At[16][136]; // stride 136 (16B aligned rows)
  int b  = blockIdx.x & 31;
  int it = blockIdx.x >> 5; // 0..7
  int lane = threadIdx.x;

  for (int x = lane; x < 16 * 128; x += 32) { // uniform trip count
    int r = x >> 7, k = x & 127;
    Qt[r][k] = PQ[((size_t)((it * 16 + r) * Bdim + b)) * 256 + 128 + k] + ba0[k];
  }
  __syncthreads();

  v16bf bfr[4];
#pragma unroll
  for (int kt = 0; kt < 4; ++kt) bfr[kt] = load_b_frag(wa1p, 1, kt, 0, lane);

  v8f acc = {};
  int row = lane >> 1, kb = (lane & 1) * 64;
  for (int j = 0; j < Ldim; ++j) {
    const float* Pj = PQ + ((size_t)(j * Bdim + b)) * 256;
#pragma unroll 8
    for (int k = 0; k < 64; k += 2) {
      float v0 = Qt[row][kb + k]     + Pj[kb + k];
      float v1 = Qt[row][kb + k + 1] + Pj[kb + k + 1];
      v0 = v0 > 0.f ? v0 : 0.f;
      v1 = v1 > 0.f ? v1 : 0.f;
      unsigned pk = (unsigned)f2bfu(v0) | ((unsigned)f2bfu(v1) << 16);
      *(unsigned*)(&At[row][kb + k]) = pk;
    }
    __syncthreads();
#pragma unroll
    for (int kt = 0; kt < 4; ++kt) {
      v16bf a = load_a_frag(&At[0][0], 136, 0, kt * 32, lane);
      acc = wmma_bf(a, bfr[kt], acc);
    }
    __syncthreads();
  }
  int n = lane & 15;
  if (n < NHdim) {
    int rbase = it * 16 + ((lane >> 4) << 3);
#pragma unroll
    for (int v = 0; v < 8; ++v)
      wout[((size_t)(rbase + v) * Bdim + b) * NHdim + n] = acc[v] + 128.f * ba1[n];
  }
}

// ---------------------------------------------------------------------------
// softmax over time axis with multiplicative mask (reference semantics:
// masked entries contribute exp(0)). One wave per (b, nh).
// ---------------------------------------------------------------------------
__global__ __launch_bounds__(256) void softmax_t(float* __restrict__ w,
                                                 const int* __restrict__ len) {
  int b = blockIdx.x;
  int nh = threadIdx.x >> 5;
  int lane = threadIdx.x & 31;
  int Lb = len[b];
  float v[4]; float mx = -1e30f;
#pragma unroll
  for (int c = 0; c < 4; ++c) {
    int l = c * 32 + lane;
    float x = w[((size_t)l * Bdim + b) * NHdim + nh];
    x = (l < Lb) ? x : 0.f;
    v[c] = x; mx = fmaxf(mx, x);
  }
#pragma unroll
  for (int o = 16; o; o >>= 1) mx = fmaxf(mx, __shfl_xor(mx, o, 32));
  float s = 0.f;
#pragma unroll
  for (int c = 0; c < 4; ++c) { v[c] = expf(v[c] - mx); s += v[c]; }
#pragma unroll
  for (int o = 16; o; o >>= 1) s += __shfl_xor(s, o, 32);
  float inv = 1.f / s;
#pragma unroll
  for (int c = 0; c < 4; ++c) {
    int l = c * 32 + lane;
    w[((size_t)l * Bdim + b) * NHdim + nh] = v[c] * inv;
  }
}

// ---------------------------------------------------------------------------
// Persistent GRU layer: one workgroup (16 wave32), h in LDS, WMMA gate GEMMs.
// Wave wv owns output columns [wv*16, wv*16+16) of H for gates r,z,n.
// Layer-0 input tile staged via CDNA5 async global->LDS DMA (ASYNCcnt).
// ---------------------------------------------------------------------------
__global__ __launch_bounds__(512) void gru_layer_k(
    int layer,
    const __bf16* __restrict__ eb,    // layer0 input embeddings [L*B][128]
    const float*  __restrict__ ts,    // [L][B][2]
    const __bf16* __restrict__ xin,   // layer1 input = out0 bf16 [L*B][256]
    const int*    __restrict__ len,
    const __bf16* __restrict__ WihP,  // packed, ntiles_tot=48
    const __bf16* __restrict__ WhhP,  // packed, ktiles=8, ntiles_tot=48
    const float* __restrict__ bih, const float* __restrict__ bhh,
    __bf16* __restrict__ out_bf,      // layer0 output (bf16)
    float*  __restrict__ out_f,       // layer1 output (f32)
    float*  __restrict__ hT) {
  __shared__ __align__(16) float  hS[32][264];
  __shared__ __align__(16) __bf16 hB[32][264];
  __shared__ __align__(16) __bf16 xB[32][168];
  __shared__ int lenS[32];
  int tid = threadIdx.x;
  int wv = tid >> 5, lane = tid & 31;
  int kxt = (layer == 0) ? 5 : 8;

  for (int x = tid; x < 32 * 264; x += 512) {
    hS[x / 264][x % 264] = 0.f;
    hB[x / 264][x % 264] = f2bf(0.f);
  }
  // zero the constant pad columns of x once (k in [128,168); 128/129 are
  // overwritten with ts each step, 130.. stay zero)
  for (int x = tid; x < 32 * 40; x += 512) {
    int r = x / 40, k = 128 + x % 40;
    xB[r][k] = f2bf(0.f);
  }
  if (tid < 32) lenS[tid] = len[tid];
  __syncthreads();

  int col = wv * 16 + (lane & 15);
  float b_r  = bih[col] + bhh[col];
  float b_z  = bih[Hdim + col] + bhh[Hdim + col];
  float bi_n = bih[2 * Hdim + col];
  float bh_n = bhh[2 * Hdim + col];
  int hi8 = (lane >> 4) << 3;

  for (int t = 0; t < Ldim; ++t) {
    if (layer == 0) {
      // ts columns (k = 128, 129) via plain stores (waves 0-1)
      if (tid < 64) {
        int r = tid >> 1, q = tid & 1;
        xB[r][Edim + q] = f2bf(ts[((size_t)(t * Bdim + r)) * 2 + q]);
      }
      // e part (k < 128): wave 0 DMAs one 256-byte row per lane, 16B chunks.
      if (wv == 0) {
        unsigned long long g = (unsigned long long)(size_t)
            (eb + (size_t)(t * Bdim + lane) * Edim);
        unsigned l32 = (unsigned)(size_t)(&xB[lane][0]); // low 32b = LDS addr
#pragma unroll
        for (int c = 0; c < 16; ++c)
          async_g2l_b128(l32 + c * 16, g + c * 16);
        wait_asynccnt0();
      }
    }
    __syncthreads(); // h + x ready

    v8f aR[2] = {}, aZ[2] = {}, aI[2] = {}, aH[2] = {};
    const __bf16* xbase = (layer == 0) ? &xB[0][0]
                                       : (xin + (size_t)(t * Bdim) * Hdim);
    int xstride = (layer == 0) ? 168 : Hdim;

    for (int kt = 0; kt < kxt; ++kt) {
      v16bf a0 = load_a_frag(xbase, xstride, 0,  kt * 32, lane);
      v16bf a1 = load_a_frag(xbase, xstride, 16, kt * 32, lane);
      v16bf br = load_b_frag(WihP, 48, kt, wv, lane);
      v16bf bz = load_b_frag(WihP, 48, kt, 16 + wv, lane);
      v16bf bn = load_b_frag(WihP, 48, kt, 32 + wv, lane);
      aR[0] = wmma_bf(a0, br, aR[0]); aR[1] = wmma_bf(a1, br, aR[1]);
      aZ[0] = wmma_bf(a0, bz, aZ[0]); aZ[1] = wmma_bf(a1, bz, aZ[1]);
      aI[0] = wmma_bf(a0, bn, aI[0]); aI[1] = wmma_bf(a1, bn, aI[1]);
    }
#pragma unroll
    for (int kt = 0; kt < 8; ++kt) {
      if (kt < 7)
        __builtin_prefetch(WhhP + ((((size_t)(kt + 1) * 48 + wv) * 32) << 4), 0, 1);
      v16bf a0 = load_a_frag(&hB[0][0], 264, 0,  kt * 32, lane);
      v16bf a1 = load_a_frag(&hB[0][0], 264, 16, kt * 32, lane);
      v16bf br = load_b_frag(WhhP, 48, kt, wv, lane);
      v16bf bz = load_b_frag(WhhP, 48, kt, 16 + wv, lane);
      v16bf bn = load_b_frag(WhhP, 48, kt, 32 + wv, lane);
      aR[0] = wmma_bf(a0, br, aR[0]); aR[1] = wmma_bf(a1, br, aR[1]);
      aZ[0] = wmma_bf(a0, bz, aZ[0]); aZ[1] = wmma_bf(a1, bz, aZ[1]);
      aH[0] = wmma_bf(a0, bn, aH[0]); aH[1] = wmma_bf(a1, bn, aH[1]);
    }
    __syncthreads(); // all reads of hB/xB complete before overwrite

#pragma unroll
    for (int m = 0; m < 2; ++m) {
#pragma unroll
      for (int v = 0; v < 8; ++v) {
        int row = m * 16 + v + hi8;
        float r = sigmf(aR[m][v] + b_r);
        float z = sigmf(aZ[m][v] + b_z);
        float n = tanhf(aI[m][v] + bi_n + r * (aH[m][v] + bh_n));
        float hold = hS[row][col];
        float hnew = (1.f - z) * n + z * hold;
        int valid = t < lenS[row];
        float hu = valid ? hnew : hold;
        float ov = valid ? hnew : 0.f;
        hS[row][col] = hu;
        hB[row][col] = f2bf(hu);
        if (layer == 0)
          out_bf[((size_t)(t * Bdim) + row) * Hdim + col] = f2bf(ov);
        else
          out_f[((size_t)(t * Bdim) + row) * Hdim + col] = ov;
      }
    }
  }
  __syncthreads();
#pragma unroll
  for (int m = 0; m < 2; ++m)
#pragma unroll
    for (int v = 0; v < 8; ++v) {
      int row = m * 16 + v + hi8;
      hT[(size_t)row * Hdim + col] = hS[row][col];
    }
}

// ---------------------------------------------------------------------------
// ctx = einsum('lbh,lbd->bhd') flattened, concatenated with hT -> zin[B][2304]
// ---------------------------------------------------------------------------
__global__ void ctx_concat(const float* __restrict__ w,
                           const float* __restrict__ out1,
                           const float* __restrict__ hT,
                           float* __restrict__ zin) {
  int idx = blockIdx.x * blockDim.x + threadIdx.x;
  if (idx >= Bdim * 2304) return;
  int b = idx / 2304, c = idx % 2304;
  float v;
  if (c < NHdim * Hdim) {
    int nh = c >> 8, hh = c & 255;
    float acc = 0.f;
    for (int l = 0; l < Ldim; ++l)
      acc += w[((size_t)l * Bdim + b) * NHdim + nh] *
             out1[((size_t)l * Bdim + b) * Hdim + hh];
    v = acc;
  } else {
    v = hT[(size_t)b * Hdim + (c - NHdim * Hdim)];
  }
  zin[idx] = v;
}

// ---------------------------------------------------------------------------
// Head: z = selu(zin @ Wf0.T + bf0); logits = z @ Wf1.T + bf1; log_softmax.
// One block per batch element.
// ---------------------------------------------------------------------------
__global__ __launch_bounds__(256) void head_k(const float* __restrict__ zin,
                                              const float* __restrict__ Wf0,
                                              const float* __restrict__ bf0,
                                              const float* __restrict__ Wf1,
                                              const float* __restrict__ bf1,
                                              float* __restrict__ out) {
  __shared__ float zb[168];
  __shared__ float lb[104];
  __shared__ float red[2];
  int b = blockIdx.x, tid = threadIdx.x;
  const float* zr = zin + (size_t)b * 2304;
  if (tid < 168) {
    float acc = bf0[tid];
    const float* wr = Wf0 + (size_t)tid * 2304;
    for (int k = 0; k < 2304; ++k) acc += zr[k] * wr[k];
    const float alpha = 1.6732632423543772f, scale = 1.0507009873554805f;
    zb[tid] = scale * (acc > 0.f ? acc : alpha * (expf(acc) - 1.f));
  }
  __syncthreads();
  if (tid < ETAdim) {
    float acc = bf1[tid];
    const float* wr = Wf1 + (size_t)tid * 168;
    for (int k = 0; k < 168; ++k) acc += zb[k] * wr[k];
    lb[tid] = acc;
  }
  __syncthreads();
  if (tid < 32) {
    float mx = -1e30f;
    for (int c = tid; c < ETAdim; c += 32) mx = fmaxf(mx, lb[c]);
#pragma unroll
    for (int o = 16; o; o >>= 1) mx = fmaxf(mx, __shfl_xor(mx, o, 32));
    float s = 0.f;
    for (int c = tid; c < ETAdim; c += 32) s += expf(lb[c] - mx);
#pragma unroll
    for (int o = 16; o; o >>= 1) s += __shfl_xor(s, o, 32);
    if (tid == 0) { red[0] = mx; red[1] = logf(s); }
  }
  __syncthreads();
  if (tid < ETAdim) out[(size_t)b * ETAdim + tid] = lb[tid] - red[0] - red[1];
}

// ---------------------------------------------------------------------------
// Launch sequence. Workspace layout (bytes, all 256-aligned), ~12.9 MB total.
// ---------------------------------------------------------------------------
extern "C" void kernel_launch(void* const* d_in, const int* in_sizes, int n_in,
                              void* d_out, int out_size, void* d_ws, size_t ws_size,
                              hipStream_t stream) {
  (void)in_sizes; (void)n_in; (void)out_size; (void)ws_size;
  const int*   node  = (const int*)d_in[0];
  const float* ts    = (const float*)d_in[1];
  const int*   lens  = (const int*)d_in[2];
  const float* table = (const float*)d_in[3];
  const float* Wih0  = (const float*)d_in[4];
  const float* Whh0  = (const float*)d_in[5];
  const float* bih0  = (const float*)d_in[6];
  const float* bhh0  = (const float*)d_in[7];
  const float* Wih1  = (const float*)d_in[8];
  const float* Whh1  = (const float*)d_in[9];
  const float* bih1  = (const float*)d_in[10];
  const float* bhh1  = (const float*)d_in[11];
  const float* Wa0   = (const float*)d_in[12];
  const float* ba0   = (const float*)d_in[13];
  const float* Wa1   = (const float*)d_in[14];
  const float* ba1   = (const float*)d_in[15];
  const float* Wf0   = (const float*)d_in[16];
  const float* bf0   = (const float*)d_in[17];
  const float* Wf1   = (const float*)d_in[18];
  const float* bf1   = (const float*)d_in[19];
  float* out = (float*)d_out;

  char* ws = (char*)d_ws;
  __bf16* e_bf  = (__bf16*)(ws + 0);         // 4096*128 bf16       = 1,048,576
  float*  PQ    = (float*) (ws + 1048576);   // 4096*256 f32        = 4,194,304
  float*  wbuf  = (float*) (ws + 5242880);   // 128*32*8 f32        =   131,072
  __bf16* out0b = (__bf16*)(ws + 5373952);   // 128*32*256 bf16     = 2,097,152
  float*  out1  = (float*) (ws + 7471104);   // 128*32*256 f32      = 4,194,304
  float*  hT    = (float*) (ws + 11665408);  // 32*256 f32          =    32,768
  float*  zin   = (float*) (ws + 11698176);  // 32*2304 f32         =   294,912
  __bf16* wa0p  = (__bf16*)(ws + 11993088);  // 4*16*512 bf16       =    65,536
  __bf16* wa1p  = (__bf16*)(ws + 12058624);  // 4*1*512 bf16        =     4,096
  __bf16* wih0p = (__bf16*)(ws + 12062720);  // 5*48*512 bf16       =   245,760
  __bf16* whh0p = (__bf16*)(ws + 12308480);  // 8*48*512 bf16       =   393,216
  __bf16* wih1p = (__bf16*)(ws + 12701696);  //                     =   393,216
  __bf16* whh1p = (__bf16*)(ws + 13094912);  //                     =   393,216

  // Weight packing (transpose + bf16 + WMMA fragment-major)
  pack_b<<<64,  256, 0, stream>>>(Wa0,  wa0p,  256, 0,   128, 128, 4, 8,  0, 16);
  pack_b<<<64,  256, 0, stream>>>(Wa0,  wa0p,  256, 128, 128, 128, 4, 8,  8, 16);
  pack_b<<<8,   256, 0, stream>>>(Wa1,  wa1p,  128, 0,   128, 8,   4, 1,  0, 1);
  pack_b<<<480, 256, 0, stream>>>(Wih0, wih0p, 130, 0,   130, 768, 5, 48, 0, 48);
  pack_b<<<768, 256, 0, stream>>>(Whh0, whh0p, 256, 0,   256, 768, 8, 48, 0, 48);
  pack_b<<<768, 256, 0, stream>>>(Wih1, wih1p, 256, 0,   256, 768, 8, 48, 0, 48);
  pack_b<<<768, 256, 0, stream>>>(Whh1, whh1p, 256, 0,   256, 768, 8, 48, 0, 48);

  embed_gather<<<2048, 256, 0, stream>>>(node, table, e_bf);
  gemm_pq<<<512, 256, 0, stream>>>(e_bf, wa0p, PQ);
  attn_w<<<256, 32, 0, stream>>>(PQ, ba0, ba1, wa1p, wbuf);
  softmax_t<<<32, 256, 0, stream>>>(wbuf, lens);

  gru_layer_k<<<1, 512, 0, stream>>>(0, e_bf, ts, (const __bf16*)nullptr, lens,
                                     wih0p, whh0p, bih0, bhh0,
                                     out0b, (float*)nullptr, hT);
  gru_layer_k<<<1, 512, 0, stream>>>(1, e_bf, ts, out0b, lens,
                                     wih1p, whh1p, bih1, bhh1,
                                     (__bf16*)nullptr, out1, hT);

  ctx_concat<<<288, 256, 0, stream>>>(wbuf, out1, hT, zin);
  head_k<<<32, 256, 0, stream>>>(zin, Wf0, bf0, Wf1, bf1, out);
}
